// DeformTransformer_6923487281683
// MI455X (gfx1250) — matrix operand order repaired and checked by reading, not compiled, not run on hardware
//
#include <hip/hip_runtime.h>
#include <hip/hip_bf16.h>
#include <cstdint>

// ---------------- problem constants ----------------
#define LEN_C   10200
#define B_C     2
#define D_C     256
#define NH_C    8
#define DH_C    32
#define DFF_C   1024
#define M_C     (B_C * LEN_C)   // 20400 rows, multiple of 16 (1275 tiles)

typedef __attribute__((ext_vector_type(16))) _Float16 v16h;
typedef __attribute__((ext_vector_type(8)))  float    v8f;

// -------- optional gfx1250 async global->LDS staging (guarded) -------------
#if defined(__has_builtin)
# if __has_builtin(__builtin_amdgcn_global_load_async_to_lds_b128)
#  define DT_ASYNC_LDS 1
# endif
#endif
#ifndef DT_ASYNC_LDS
# define DT_ASYNC_LDS 0
#endif

#if DT_ASYNC_LDS
typedef int dt_v4i __attribute__((vector_size(16)));
typedef __attribute__((address_space(1))) dt_v4i dt_gv4i;
typedef __attribute__((address_space(3))) dt_v4i dt_lv4i;
#endif

static __device__ __forceinline__ void dt_cp16(void* lds_dst, const void* g_src) {
#if DT_ASYNC_LDS
    __builtin_amdgcn_global_load_async_to_lds_b128(
        (dt_gv4i*)(uintptr_t)g_src,
        (dt_lv4i*)(uint32_t)(uintptr_t)lds_dst, 0, 0);
#else
    *(uint4*)lds_dst = *(const uint4*)g_src;   // global_load_b128 + ds_store_b128
#endif
}

static __device__ __forceinline__ void dt_stage_fence() {
#if DT_ASYNC_LDS
# if __has_builtin(__builtin_amdgcn_s_wait_asynccnt)
    __builtin_amdgcn_s_wait_asynccnt(0);
# else
    asm volatile("s_wait_asynccnt 0x0" ::: "memory");
# endif
#endif
}

// ---------------- weight convert + transpose: WT[n*K+k] = (f16)W[k*N+n] ----
__global__ void dt_convwt_kernel(const float* __restrict__ W, _Float16* __restrict__ WT,
                                 int K, int N) {
    int i = blockIdx.x * blockDim.x + threadIdx.x;
    if (i < K * N) {
        int k = i / N, n = i % N;
        WT[(size_t)n * K + k] = (_Float16)W[i];
    }
}

// ---------------- prep: q16 = f16(x+pos), x16 = f16(x) ---------------------
__global__ void dt_prep_kernel(const float* __restrict__ x, const float* __restrict__ pos,
                               _Float16* __restrict__ q16, _Float16* __restrict__ x16, int n) {
    int i = blockIdx.x * blockDim.x + threadIdx.x;
    if (i < n) {
        float xv = x[i];
        q16[i] = (_Float16)(xv + pos[i]);
        x16[i] = (_Float16)xv;
    }
}

// ---------------- LDS-staged WMMA GEMM -------------------------------------
// C[M,N] = A16[M,K] @ WT16[N,K]^T + bias.  Block = 256 threads (8 waves).
// Block tile: 16 (M) x 128 (N); per 32-wide K chunk the block stages
// A[16x32] and B[128x32] in LDS; wave w computes the 16x16 tile at n-sub w*16.
__global__ __launch_bounds__(256)
void dt_wmma_gemm_kernel(const _Float16* __restrict__ A, const _Float16* __restrict__ WT,
                         const float* __restrict__ bias, float* __restrict__ C,
                         _Float16* __restrict__ C16, int M, int N, int K, int relu) {
    __shared__ _Float16 Alds[16][32];    // 1 KB
    __shared__ _Float16 Blds[128][32];   // 8 KB

    int tid  = threadIdx.x;
    int wave = tid >> 5;
    int lane = tid & 31;
    int mr   = lane & 15;
    int half = lane >> 4;
    int m0 = blockIdx.x << 4;
    int n0 = blockIdx.y << 7;

    v8f c = {};
    for (int k0 = 0; k0 < K; k0 += 32) {
        __syncthreads();   // previous iteration's LDS readers done
        // stage A[16x32]: threads 0..63, each one 16B granule
        if (tid < 64) {
            int row = tid >> 2, part = tid & 3;           // part: 8 halfs
            dt_cp16(&Alds[row][part * 8],
                    A + (size_t)(m0 + row) * K + k0 + part * 8);
        }
        // stage B[128x32]: each thread two 16B granules (one 32B half-row)
        {
            int row = tid >> 1, part = tid & 1;           // part: 16 halfs
            const _Float16* g = WT + (size_t)(n0 + row) * K + k0 + part * 16;
            dt_cp16(&Blds[row][part * 16], g);
            dt_cp16(&Blds[row][part * 16 + 8], g + 8);
        }
        if (k0 + 32 < K && tid < 64) {
            int row = tid >> 2, part = tid & 3;
            __builtin_prefetch(A + (size_t)(m0 + row) * K + k0 + 32 + part * 8, 0, 0);
        }
        dt_stage_fence();
        __syncthreads();

        // build fragments from LDS (contiguous per-lane runs -> ds_load_b128)
        v16h a, b;
        const _Float16* Ar = &Alds[mr][half * 8];
#pragma unroll
        for (int i = 0; i < 8; ++i) {
            a[i]     = Ar[i];        // K = half*8 + i
            a[8 + i] = Ar[16 + i];   // K = 16 + half*8 + i
        }
        const _Float16* Br = &Blds[(wave << 4) + mr][half * 8];
#pragma unroll
        for (int i = 0; i < 8; ++i) {
            b[i]     = Br[i];        // K = half*8 + i
            b[8 + i] = Br[16 + i];   // K = 16 + half*8 + i
        }

        c = __builtin_amdgcn_wmma_f32_16x16x32_f16(false, a, false, b, (short)0, c,
                                                   false, false);
    }

    int ncol = n0 + (wave << 4) + mr;
    float bv = bias[ncol];
#pragma unroll
    for (int j = 0; j < 8; ++j) {
        int row = m0 + (half << 3) + j;   // C: VGPR j = row half*8+j, col = lane&15
        float v = c[j] + bv;
        if (relu) v = fmaxf(v, 0.0f);
        size_t idx = (size_t)row * N + ncol;
        C[idx] = v;
        if (C16) C16[idx] = (_Float16)v;
    }
}

// ---------------- deformable attention sampling ----------------------------
// one block per (b,q); thread t -> head h = t/32, channel dh = t%32
__global__ __launch_bounds__(256)
void dt_sample_kernel(const float* __restrict__ off, const float* __restrict__ logits,
                      const float* __restrict__ val, const float* __restrict__ vr,
                      float* __restrict__ out, _Float16* __restrict__ out16) {
    const int LVL_H[4] = {48, 24, 12, 6};
    const int LVL_W[4] = {160, 80, 40, 20};
    const int LVL_S[4] = {0, 7680, 9600, 10080};

    int bq = blockIdx.x;
    int b = bq / LEN_C, q = bq % LEN_C;
    int t = threadIdx.x;
    int h = t >> 5, dh = t & 31;

    // reference point base from query's own level
    int lvl0 = (q < 7680) ? 0 : (q < 9600) ? 1 : (q < 10080) ? 2 : 3;
    int qi = q - LVL_S[lvl0];
    int row = qi / LVL_W[lvl0], col = qi % LVL_W[lvl0];
    float rxb = (col + 0.5f) / (vr[b * 8 + lvl0 * 2 + 0] * (float)LVL_W[lvl0]);
    float ryb = (row + 0.5f) / (vr[b * 8 + lvl0 * 2 + 1] * (float)LVL_H[lvl0]);

    // softmax over 16 attention logits for this (q,h)
    const float* L = logits + (size_t)bq * 128 + h * 16;
    float mx = -1e30f;
#pragma unroll
    for (int i = 0; i < 16; ++i) mx = fmaxf(mx, L[i]);
    float w[16], sum = 0.0f;
#pragma unroll
    for (int i = 0; i < 16; ++i) { w[i] = expf(L[i] - mx); sum += w[i]; }
    float inv = 1.0f / sum;

    const float* offr = off + (size_t)bq * 256 + h * 32;  // ((h*4+lvl)*4+p)*2+c
    float acc = 0.0f;
#pragma unroll
    for (int lvl = 0; lvl < 4; ++lvl) {
        int Hl = LVL_H[lvl], Wl = LVL_W[lvl];
        float vx = vr[b * 8 + lvl * 2 + 0], vy = vr[b * 8 + lvl * 2 + 1];
        const float* vb = val + ((size_t)(b * LEN_C + LVL_S[lvl])) * 256 + h * 32 + dh;
#pragma unroll
        for (int p = 0; p < 4; ++p) {
            float ox = offr[(lvl * 4 + p) * 2 + 0];
            float oy = offr[(lvl * 4 + p) * 2 + 1];
            float lx = rxb * vx + ox / (float)Wl;
            float ly = ryb * vy + oy / (float)Hl;
            float xf = lx * (float)Wl - 0.5f;
            float yf = ly * (float)Hl - 0.5f;
            float x0f = floorf(xf), y0f = floorf(yf);
            float fx = xf - x0f, fy = yf - y0f;
            int x0 = (int)x0f, y0 = (int)y0f;
            float aw = w[lvl * 4 + p] * inv;
#pragma unroll
            for (int dy = 0; dy < 2; ++dy)
#pragma unroll
                for (int dx = 0; dx < 2; ++dx) {
                    int xi = x0 + dx, yi = y0 + dy;
                    if (xi >= 0 && xi < Wl && yi >= 0 && yi < Hl) {
                        float ww = (dx ? fx : 1.0f - fx) * (dy ? fy : 1.0f - fy);
                        acc += aw * ww * vb[((size_t)yi * Wl + xi) * 256];
                    }
                }
        }
    }
    size_t oi = (size_t)bq * 256 + h * 32 + dh;
    out[oi] = acc;
    out16[oi] = (_Float16)acc;
}

// ---------------- fused residual + layernorm (D=256) -----------------------
__global__ __launch_bounds__(256)
void dt_addln_kernel(const float* __restrict__ xin, const float* __restrict__ delta,
                     const float* __restrict__ g, const float* __restrict__ beta,
                     float* __restrict__ xout, _Float16* __restrict__ xout16) {
    __shared__ float red[256];
    int r = blockIdx.x, t = threadIdx.x;
    size_t idx = (size_t)r * 256 + t;
    float v = xin[idx] + delta[idx];
    red[t] = v;
    __syncthreads();
    for (int s = 128; s > 0; s >>= 1) { if (t < s) red[t] += red[t + s]; __syncthreads(); }
    float mean = red[0] * (1.0f / 256.0f);
    __syncthreads();
    float d = v - mean;
    red[t] = d * d;
    __syncthreads();
    for (int s = 128; s > 0; s >>= 1) { if (t < s) red[t] += red[t + s]; __syncthreads(); }
    float var = red[0] * (1.0f / 256.0f);
    float o = d * rsqrtf(var + 1e-5f) * g[t] + beta[t];
    xout[idx] = o;
    if (xout16) xout16[idx] = (_Float16)o;
}

// ---------------- host orchestration ---------------------------------------
extern "C" void kernel_launch(void* const* d_in, const int* in_sizes, int n_in,
                              void* d_out, int out_size, void* d_ws, size_t ws_size,
                              hipStream_t stream) {
    const float* src    = (const float*)d_in[0];
    const float* pos    = (const float*)d_in[1];
    const float* vr     = (const float*)d_in[2];
    const float* W_off  = (const float*)d_in[3];
    const float* b_off  = (const float*)d_in[4];
    const float* W_attn = (const float*)d_in[5];
    const float* b_attn = (const float*)d_in[6];
    const float* W_val  = (const float*)d_in[7];
    const float* b_val  = (const float*)d_in[8];
    const float* W_out  = (const float*)d_in[9];
    const float* b_out  = (const float*)d_in[10];
    const float* ln1g   = (const float*)d_in[11];
    const float* ln1b   = (const float*)d_in[12];
    const float* W_ff1  = (const float*)d_in[13];
    const float* b_ff1  = (const float*)d_in[14];
    const float* W_ff2  = (const float*)d_in[15];
    const float* b_ff2  = (const float*)d_in[16];
    const float* ln2g   = (const float*)d_in[17];
    const float* ln2b   = (const float*)d_in[18];

    const size_t M = M_C;
    char* ws = (char*)d_ws;
    auto bump = [&](size_t bytes) -> char* {
        char* p = ws;
        ws += (bytes + 255) & ~(size_t)255;
        return p;
    };

    float*    xbuf  = (float*)bump(M * 256 * 4);       // persistent activations
    _Float16* q16   = (_Float16*)bump(M * 256 * 2);
    _Float16* x16   = (_Float16*)bump(M * 256 * 2);
    // big region: attention-stage buffers alias FFN hidden buffer (both 83.56 MB)
    char* Dreg = bump(M * 1024 * 4);
    float*    offb  = (float*)(Dreg);
    float*    logb  = (float*)(Dreg + M * 256 * 4);
    float*    valb  = (float*)(Dreg + M * 256 * 4 + M * 128 * 4);
    float*    attb  = (float*)(Dreg + M * 256 * 4 + M * 128 * 4 + M * 256 * 4);
    _Float16* att16 = (_Float16*)(Dreg + M * 256 * 4 + M * 128 * 4 + 2 * M * 256 * 4);
    float*    hbuf  = (float*)Dreg;                    // FFN hidden (after attn stage done)
    _Float16* h16   = (_Float16*)bump(M * 1024 * 2);
    float*    projb = (float*)bump(M * 256 * 4);       // attn projection / ffn output
    _Float16* wt_off  = (_Float16*)bump(256 * 256 * 2);
    _Float16* wt_attn = (_Float16*)bump(256 * 128 * 2);
    _Float16* wt_val  = (_Float16*)bump(256 * 256 * 2);
    _Float16* wt_out  = (_Float16*)bump(256 * 256 * 2);
    _Float16* wt_ff1  = (_Float16*)bump(256 * 1024 * 2);
    _Float16* wt_ff2  = (_Float16*)bump(1024 * 256 * 2);

    const float* xcur = src;
    for (int l = 0; l < 3; ++l) {
        // convert+transpose this layer's weights to f16
        dt_convwt_kernel<<<(256 * 256 + 255) / 256, 256, 0, stream>>>(W_off  + (size_t)l * 256 * 256,  wt_off,  256, 256);
        dt_convwt_kernel<<<(256 * 128 + 255) / 256, 256, 0, stream>>>(W_attn + (size_t)l * 256 * 128,  wt_attn, 256, 128);
        dt_convwt_kernel<<<(256 * 256 + 255) / 256, 256, 0, stream>>>(W_val  + (size_t)l * 256 * 256,  wt_val,  256, 256);
        dt_convwt_kernel<<<(256 * 256 + 255) / 256, 256, 0, stream>>>(W_out  + (size_t)l * 256 * 256,  wt_out,  256, 256);
        dt_convwt_kernel<<<(256 * 1024 + 255) / 256, 256, 0, stream>>>(W_ff1 + (size_t)l * 256 * 1024, wt_ff1,  256, 1024);
        dt_convwt_kernel<<<(1024 * 256 + 255) / 256, 256, 0, stream>>>(W_ff2 + (size_t)l * 1024 * 256, wt_ff2,  1024, 256);

        // q = x + pos (f16), x (f16)
        int nEl = (int)(M * 256);
        dt_prep_kernel<<<(nEl + 255) / 256, 256, 0, stream>>>(xcur, pos, q16, x16, nEl);

        // sampling offsets, attention logits, value projection
        dt_wmma_gemm_kernel<<<dim3(M / 16, 256 / 128), 256, 0, stream>>>(q16, wt_off,  b_off  + l * 256, offb, (_Float16*)nullptr, (int)M, 256, 256, 0);
        dt_wmma_gemm_kernel<<<dim3(M / 16, 128 / 128), 256, 0, stream>>>(q16, wt_attn, b_attn + l * 128, logb, (_Float16*)nullptr, (int)M, 128, 256, 0);
        dt_wmma_gemm_kernel<<<dim3(M / 16, 256 / 128), 256, 0, stream>>>(x16, wt_val,  b_val  + l * 256, valb, (_Float16*)nullptr, (int)M, 256, 256, 0);

        // deformable attention gather
        dt_sample_kernel<<<(int)M, 256, 0, stream>>>(offb, logb, valb, vr, attb, att16);

        // output projection, residual + LN1
        dt_wmma_gemm_kernel<<<dim3(M / 16, 256 / 128), 256, 0, stream>>>(att16, wt_out, b_out + l * 256, projb, (_Float16*)nullptr, (int)M, 256, 256, 0);
        dt_addln_kernel<<<(int)M, 256, 0, stream>>>(xcur, projb, ln1g + l * 256, ln1b + l * 256, xbuf, x16);

        // FFN: relu(x @ Wff1 + b) @ Wff2 + b, residual + LN2
        dt_wmma_gemm_kernel<<<dim3(M / 16, 1024 / 128), 256, 0, stream>>>(x16, wt_ff1, b_ff1 + l * 1024, hbuf, h16, (int)M, 1024, 256, 1);
        dt_wmma_gemm_kernel<<<dim3(M / 16, 256 / 128), 256, 0, stream>>>(h16, wt_ff2, b_ff2 + l * 256, projb, (_Float16*)nullptr, (int)M, 256, 1024, 0);
        float* lnOut = (l == 2) ? (float*)d_out : xbuf;
        dt_addln_kernel<<<(int)M, 256, 0, stream>>>(xbuf, projb, ln2g + l * 256, ln2b + l * 256, lnOut, x16);

        xcur = xbuf;
    }
}
